// IMHA_64141041598708
// MI455X (gfx1250) — matrix operand order repaired and checked by reading, not compiled, not run on hardware
//
#include <hip/hip_runtime.h>

#define Bsz 2048
#define Nf  128
#define Dd  32
#define Hh  8
#define NDk 4096
#define Uu  512

typedef __attribute__((ext_vector_type(16))) __bf16        v16bf;
typedef __attribute__((ext_vector_type(2)))  __bf16        v2bf;
typedef __attribute__((ext_vector_type(8)))  float         v8f;
typedef __attribute__((ext_vector_type(8)))  unsigned int  v8u;
typedef __attribute__((ext_vector_type(4)))  unsigned int  v4u;
typedef __attribute__((ext_vector_type(4)))  float         v4f;

static __device__ __forceinline__ unsigned short f2bf(float f) {
  return __builtin_bit_cast(unsigned short, (__bf16)f);  // native v_cvt, RNE
}
static __device__ __forceinline__ unsigned int pack2bf(float lo, float hi) {
#if __has_builtin(__builtin_amdgcn_cvt_pk_bf16_f32)
  return __builtin_bit_cast(unsigned int,
                            __builtin_amdgcn_cvt_pk_bf16_f32(lo, hi));
#else
  // build_vector of two fptruncs -> single v_cvt_pk_bf16_f32
  v2bf p = {(__bf16)lo, (__bf16)hi};
  return __builtin_bit_cast(unsigned int, p);
#endif
}
static __device__ __forceinline__ v8u joinu(v4u lo, v4u hi) {
  v8u r = {lo[0], lo[1], lo[2], lo[3], hi[0], hi[1], hi[2], hi[3]};
  return r;
}
static __device__ __forceinline__ v8f zero8() {
  v8f z = {0.f, 0.f, 0.f, 0.f, 0.f, 0.f, 0.f, 0.f};
  return z;
}
static __device__ __forceinline__ v8f wmma_bf16(v8u a, v8u b, v8f c) {
  return __builtin_amdgcn_wmma_f32_16x16x32_bf16(
      false, __builtin_bit_cast(v16bf, a),
      false, __builtin_bit_cast(v16bf, b),
      (short)0, c, false, false);
}

// ---------------------------------------------------------------------------
// Stage 1: per-head projection GEMM  Out[h] (B x U) = X (B x ND) * W[h] (ND x U)
// f32 -> bf16 on the fly; double-buffered LDS + software-pipelined global loads
// (one barrier per k-step; next tile's loads overlap current tile's WMMAs).
// transposeOut=1 stores Out as (U x B) per head (so attention's PV staging is a
// pure 2-D tile copy / TDM descriptor) and uses packed b128 stores.
// ---------------------------------------------------------------------------
__global__ __launch_bounds__(256) void proj_gemm(const float* __restrict__ X,
                                                 const float* __restrict__ W,
                                                 unsigned short* __restrict__ Out,
                                                 int transposeOut) {
  const int h  = blockIdx.z;
  const int m0 = blockIdx.x * 128;
  const int n0 = blockIdx.y * 64;
  const float* Wh = W + (size_t)h * NDk * Uu;
  unsigned short* Oh = Out + (size_t)h * Bsz * Uu;

  __shared__ unsigned int lA[2][128 * 16];  // 2 x (128 rows x 32 k, bf16 pairs)
  __shared__ unsigned int lB[2][64 * 16];   // 2 x (64 n x 32 k, k-pairs per n)

  const int tid  = threadIdx.x;
  const int wv   = tid >> 5;
  const int lane = tid & 31;
  const int ln   = lane & 15;
  const int hb   = lane >> 4;
  const int wm   = (wv >> 1) * 32;
  const int wn   = (wv & 1) * 32;

  const int rowA = tid >> 1;            // A-stage role
  const int khA  = (tid & 1) * 16;
  const int kpB  = tid >> 4;            // B-stage role (k-pair)
  const int nbB  = (tid & 15) * 4;

  v4f ra0, ra1, ra2, ra3, rbl, rbh;
  auto load_next = [&](int k0) {
    const float* srcA = X + (size_t)(m0 + rowA) * NDk + (k0 + khA);
    __builtin_prefetch(srcA + 64, 0, 1);
    ra0 = *(const v4f*)(srcA);
    ra1 = *(const v4f*)(srcA + 4);
    ra2 = *(const v4f*)(srcA + 8);
    ra3 = *(const v4f*)(srcA + 12);
    const float* srcB = Wh + (size_t)(k0 + 2 * kpB) * Uu + (n0 + nbB);
    rbl = *(const v4f*)(srcB);
    rbh = *(const v4f*)(srcB + Uu);
  };

  v8f acc[2][2];
#pragma unroll
  for (int i = 0; i < 2; ++i)
#pragma unroll
    for (int j = 0; j < 2; ++j) acc[i][j] = zero8();

  load_next(0);
  int pb = 0;
#pragma unroll 1
  for (int k0 = 0; k0 < NDk; k0 += 32) {
    // pack staged registers into LDS buffer pb
    {
      unsigned int* dst = &lA[pb][rowA * 16 + (khA >> 1)];
      dst[0] = pack2bf(ra0[0], ra0[1]); dst[1] = pack2bf(ra0[2], ra0[3]);
      dst[2] = pack2bf(ra1[0], ra1[1]); dst[3] = pack2bf(ra1[2], ra1[3]);
      dst[4] = pack2bf(ra2[0], ra2[1]); dst[5] = pack2bf(ra2[2], ra2[3]);
      dst[6] = pack2bf(ra3[0], ra3[1]); dst[7] = pack2bf(ra3[2], ra3[3]);
#pragma unroll
      for (int j = 0; j < 4; ++j)
        lB[pb][(nbB + j) * 16 + kpB] = pack2bf(rbl[j], rbh[j]);
    }
    __syncthreads();
    if (k0 + 32 < NDk) load_next(k0 + 32);  // overlap with WMMAs below

    v8u af[2], bfv[2];
#pragma unroll
    for (int mi = 0; mi < 2; ++mi) {
      const unsigned int* p = &lA[pb][(wm + mi * 16 + ln) * 16];
      af[mi] = joinu(*(const v4u*)(p + hb * 4), *(const v4u*)(p + hb * 4 + 8));
    }
#pragma unroll
    for (int nj = 0; nj < 2; ++nj) {
      const unsigned int* p = &lB[pb][(wn + nj * 16 + ln) * 16];
      bfv[nj] = joinu(*(const v4u*)(p + hb * 4), *(const v4u*)(p + hb * 4 + 8));
    }
#pragma unroll
    for (int mi = 0; mi < 2; ++mi)
#pragma unroll
      for (int nj = 0; nj < 2; ++nj)
        acc[mi][nj] = wmma_bf16(af[mi], bfv[nj], acc[mi][nj]);
    pb ^= 1;
  }

  if (transposeOut) {
    // lane's 8 acc elems = 8 consecutive m at fixed n -> one b128 store each
#pragma unroll
    for (int mi = 0; mi < 2; ++mi)
#pragma unroll
      for (int nj = 0; nj < 2; ++nj) {
        int n = n0 + wn + nj * 16 + ln;
        int m = m0 + wm + mi * 16 + hb * 8;
        v4u pk = {pack2bf(acc[mi][nj][0], acc[mi][nj][1]),
                  pack2bf(acc[mi][nj][2], acc[mi][nj][3]),
                  pack2bf(acc[mi][nj][4], acc[mi][nj][5]),
                  pack2bf(acc[mi][nj][6], acc[mi][nj][7])};
        *(v4u*)(Oh + (size_t)n * Bsz + m) = pk;
      }
  } else {
#pragma unroll
    for (int mi = 0; mi < 2; ++mi)
#pragma unroll
      for (int nj = 0; nj < 2; ++nj)
#pragma unroll
        for (int r = 0; r < 8; ++r) {
          int m = m0 + wm + mi * 16 + r + hb * 8;
          int n = n0 + wn + nj * 16 + ln;
          Oh[(size_t)m * Uu + n] = f2bf(acc[mi][nj][r]);
        }
  }
}

// ---------------------------------------------------------------------------
// Stage 2: flash-style attention per head. Q,K: (H,B,U) bf16; Vt: (H,U,B) bf16.
// Block = 64 query rows x 1 head; wave = 16 q-rows x 256-wide U-half.
// Q fragments hoisted to registers (loaded once); V chunk (512 u x 32 keys,
// 4 KB row stride) staged into LDS by the Tensor Data Mover when available.
// ---------------------------------------------------------------------------
__global__ __launch_bounds__(256) void attn(const unsigned short* __restrict__ Q,
                                            const unsigned short* __restrict__ K,
                                            const unsigned short* __restrict__ Vt,
                                            float* __restrict__ Concat) {
  const int h     = blockIdx.y;
  const int tid   = threadIdx.x;
  const int wv    = tid >> 5;
  const int lane  = tid & 31;
  const int ln    = lane & 15;
  const int hb    = lane >> 4;
  const int qgrp  = wv >> 1;
  const int uhalf = wv & 1;
  const int qbase = blockIdx.x * 64 + qgrp * 16;

  const unsigned short* Qh = Q  + (size_t)h * Bsz * Uu;
  const unsigned short* Kh = K  + (size_t)h * Bsz * Uu;
  const unsigned short* Vh = Vt + (size_t)h * Bsz * Uu;  // (U x B)

  __shared__ unsigned int   sV[Uu * 16];      // 512 u-rows x 32 keys (32 KB)
  __shared__ unsigned short sPu[8][16 * 32];  // per-wave P scratch     (8 KB)

  v8f Oacc[16];
#pragma unroll
  for (int t = 0; t < 16; ++t) Oacc[t] = zero8();
  float rm[8], rl[8];
#pragma unroll
  for (int r = 0; r < 8; ++r) { rm[r] = -1e30f; rl[r] = 0.f; }

  const float scale = 0.17677669529663687f;  // 1/sqrt(D), D=32

  // ---- hoist this wave's 16x512 Q tile into registers (read once from HBM)
  v8u qfr[16];
  {
    const unsigned int* qrow =
        (const unsigned int*)(Qh + (size_t)(qbase + ln) * Uu);
#pragma unroll
    for (int t = 0; t < 16; ++t) {
      const unsigned int* qp = qrow + t * 16;
      qfr[t] = joinu(*(const v4u*)(qp + hb * 4), *(const v4u*)(qp + hb * 4 + 8));
    }
  }

#if __has_builtin(__builtin_amdgcn_tensor_load_to_lds)
  typedef __attribute__((ext_vector_type(4))) unsigned int u32x4;
  typedef __attribute__((ext_vector_type(4))) int          i32x4;
  typedef __attribute__((ext_vector_type(8))) int          i32x8;
  const unsigned lds_base = (unsigned)(size_t)(void*)&sV[0];
  // D# group1: wg_mask=0, data_size=2B; tensor_dim0=2048, dim0_stride=2048;
  // tensor_dim1=512; tile_dim0=32 keys, tile_dim1=512 u-rows.
  const i32x8 g1 = {(int)0x00010000u,   // [17:16] data_size=1 (2 bytes)
                    (int)0x08000000u,   // tensor_dim0[15:0]=2048 in [31:16]
                    (int)0x02000000u,   // tensor_dim1[15:0]=512 in [31:16]
                    (int)0x00200000u,   // tile_dim0=32 in [31:16]
                    (int)0x00000200u,   // tile_dim1=512
                    (int)2048,          // tensor_dim0_stride[31:0]
                    0, 0};
  const i32x4 gz = {0, 0, 0, 0};
#if __clang_major__ >= 23
  const i32x8 gz8 = {0, 0, 0, 0, 0, 0, 0, 0};
#endif
#endif

#pragma unroll 1
  for (int c0 = 0; c0 < Bsz; c0 += 32) {
    __syncthreads();  // all readers of sV done
#if __has_builtin(__builtin_amdgcn_tensor_load_to_lds)
    if (wv == 0) {
      unsigned long long ga = (unsigned long long)(size_t)(Vh + c0);
      u32x4 g0 = {1u,                                  // count=1 valid D#
                  lds_base,                            // lds_addr
                  (unsigned)(ga & 0xffffffffu),        // global_addr[31:0]
                  (unsigned)((ga >> 32) & 0x1ffffffu) | 0x80000000u};  // +type=2
#if __clang_major__ >= 23
      __builtin_amdgcn_tensor_load_to_lds(g0, g1, gz, gz, gz8, 0);
#else
      __builtin_amdgcn_tensor_load_to_lds(g0, g1, gz, gz, 0);
#endif
      __builtin_amdgcn_s_wait_tensorcnt((short)0);
    }
#else
    {
      int u2 = tid * 2;  // 512 rows / 256 threads
#pragma unroll
      for (int rr = 0; rr < 2; ++rr) {
        const v4u* src = (const v4u*)(Vh + (size_t)(u2 + rr) * Bsz + c0);
        v4u* dst = (v4u*)&sV[(u2 + rr) * 16];
        dst[0] = src[0]; dst[1] = src[1]; dst[2] = src[2]; dst[3] = src[3];
      }
    }
#endif
    __syncthreads();

    // ---- S = Q Kt for this 16q x 32key chunk (Q from registers)
    v8f s0 = zero8(), s1 = zero8();
    const unsigned int* krow0 =
        (const unsigned int*)(Kh + (size_t)(c0 + ln) * Uu);
    const unsigned int* krow1 =
        (const unsigned int*)(Kh + (size_t)(c0 + 16 + ln) * Uu);
#pragma unroll
    for (int t = 0; t < 16; ++t) {
      const unsigned int* kp0 = krow0 + t * 16;
      const unsigned int* kp1 = krow1 + t * 16;
      v8u kf0 = joinu(*(const v4u*)(kp0 + hb * 4), *(const v4u*)(kp0 + hb * 4 + 8));
      v8u kf1 = joinu(*(const v4u*)(kp1 + hb * 4), *(const v4u*)(kp1 + hb * 4 + 8));
      s0 = wmma_bf16(qfr[t], kf0, s0);
      s1 = wmma_bf16(qfr[t], kf1, s1);
    }

    // ---- online softmax (row reductions within 16-lane halves)
    float fac[8];
#pragma unroll
    for (int r = 0; r < 8; ++r) {
      float a = s0[r] * scale, b = s1[r] * scale;
      float cm = fmaxf(a, b);
      cm = fmaxf(cm, __shfl_xor(cm, 1, 32));
      cm = fmaxf(cm, __shfl_xor(cm, 2, 32));
      cm = fmaxf(cm, __shfl_xor(cm, 4, 32));
      cm = fmaxf(cm, __shfl_xor(cm, 8, 32));
      float nm = fmaxf(rm[r], cm);
      float f  = __expf(rm[r] - nm);
      float p0 = __expf(a - nm);
      float p1 = __expf(b - nm);
      float rs = p0 + p1;
      rs += __shfl_xor(rs, 1, 32);
      rs += __shfl_xor(rs, 2, 32);
      rs += __shfl_xor(rs, 4, 32);
      rs += __shfl_xor(rs, 8, 32);
      rl[r] = rl[r] * f + rs;
      rm[r] = nm;
      fac[r] = f;
      s0[r] = p0;
      s1[r] = p1;
    }
#pragma unroll
    for (int t = 0; t < 16; ++t)
#pragma unroll
      for (int r = 0; r < 8; ++r) Oacc[t][r] *= fac[r];

    // ---- bounce P through LDS to re-layout acc -> A fragment (bf16)
#pragma unroll
    for (int r = 0; r < 8; ++r) {
      int m = r + hb * 8;
      sPu[wv][m * 32 + ln]      = f2bf(s0[r]);
      sPu[wv][m * 32 + 16 + ln] = f2bf(s1[r]);
    }
    const unsigned int* pp = (const unsigned int*)&sPu[wv][ln * 32];
    v8u pf = joinu(*(const v4u*)(pp + hb * 4), *(const v4u*)(pp + hb * 4 + 8));

    // ---- O += P * V over this wave's 256-wide U half
#pragma unroll
    for (int t = 0; t < 16; ++t) {
      int u0 = uhalf * 256 + t * 16;
      const unsigned int* vp = &sV[(u0 + ln) * 16 + hb * 4];
      v8u vf = joinu(*(const v4u*)(vp), *(const v4u*)(vp + 8));
      Oacc[t] = wmma_bf16(pf, vf, Oacc[t]);
    }
  }

  // ---- normalize + scatter into concat layout (b, n, h*4+j)
  float inv[8];
#pragma unroll
  for (int r = 0; r < 8; ++r) inv[r] = 1.f / rl[r];
#pragma unroll
  for (int t = 0; t < 16; ++t) {
    int u = uhalf * 256 + t * 16 + ln;
    int nfeat = u >> 2, j = u & 3;
#pragma unroll
    for (int r = 0; r < 8; ++r) {
      int q = qbase + r + hb * 8;
      Concat[((size_t)q * Nf + nfeat) * Dd + h * 4 + j] = Oacc[t][r] * inv[r];
    }
  }
}

// ---------------------------------------------------------------------------
// Stage 3: Y (B*N x 32) = Concat (B*N x 32) * Wo (32 x 32), K=32 -> 1 wmma/tile
// ---------------------------------------------------------------------------
__global__ __launch_bounds__(256) void out_proj(const float* __restrict__ C,
                                                const float* __restrict__ Wo,
                                                float* __restrict__ Y) {
  const int tid  = threadIdx.x;
  const int wv   = tid >> 5;
  const int lane = tid & 31;
  const int ln   = lane & 15;
  const int hb   = lane >> 4;
  const size_t r0 = (size_t)blockIdx.x * 128 + (size_t)wv * 16;

  const float* row = C + (r0 + ln) * Dd;
  unsigned int a[8];
#pragma unroll
  for (int d = 0; d < 8; ++d) {
    int kp = hb * 8 + d;
    a[d] = pack2bf(row[2 * kp], row[2 * kp + 1]);
  }
  v8u afrag = {a[0], a[1], a[2], a[3], a[4], a[5], a[6], a[7]};

  v8f acc[2];
#pragma unroll
  for (int nt = 0; nt < 2; ++nt) {
    unsigned int b[8];
#pragma unroll
    for (int d = 0; d < 8; ++d)
      b[d] = pack2bf(Wo[(hb * 16 + 2 * d) * Dd + nt * 16 + ln],
                     Wo[(hb * 16 + 2 * d + 1) * Dd + nt * 16 + ln]);
    v8u bfrag = {b[0], b[1], b[2], b[3], b[4], b[5], b[6], b[7]};
    acc[nt] = wmma_bf16(afrag, bfrag, zero8());
  }

#pragma unroll
  for (int nt = 0; nt < 2; ++nt)
#pragma unroll
    for (int r = 0; r < 8; ++r) {
      size_t m = r0 + (size_t)(r + hb * 8);
      Y[m * Dd + nt * 16 + ln] = acc[nt][r];
    }
}

// ---------------------------------------------------------------------------
extern "C" void kernel_launch(void* const* d_in, const int* in_sizes, int n_in,
                              void* d_out, int out_size, void* d_ws, size_t ws_size,
                              hipStream_t stream) {
  (void)in_sizes; (void)n_in; (void)out_size; (void)ws_size;
  const float* x  = (const float*)d_in[0];
  const float* Wq = (const float*)d_in[1];
  const float* Wk = (const float*)d_in[2];
  const float* Wv = (const float*)d_in[3];
  const float* Wo = (const float*)d_in[4];
  float* out = (float*)d_out;

  const size_t pe = (size_t)Hh * Bsz * Uu;  // elems per bf16 q/k/v tensor
  unsigned short* qb = (unsigned short*)d_ws;
  unsigned short* kb = qb + pe;
  unsigned short* vb = kb + pe;   // stored transposed: (H, U, B)
  float* concat = (float*)(vb + pe);

  dim3 g1(Bsz / 128, Uu / 64, Hh);  // (16, 8, 8)
  proj_gemm<<<g1, 256, 0, stream>>>(x, Wq, qb, 0);
  proj_gemm<<<g1, 256, 0, stream>>>(x, Wk, kb, 0);
  proj_gemm<<<g1, 256, 0, stream>>>(x, Wv, vb, 1);
  attn<<<dim3(Bsz / 64, Hh), 256, 0, stream>>>(qb, kb, vb, concat);
  out_proj<<<dim3((Bsz * Nf) / 128), 256, 0, stream>>>(concat, Wo, out);
}
